// BackgroundEncoder_15178414424817
// MI455X (gfx1250) — matrix operand order repaired and checked by reading, not compiled, not run on hardware
//
#include <hip/hip_runtime.h>

// ---------------------------------------------------------------------------
// CDNA5 (gfx1250, wave32) implementation of BackgroundEncoder forward.
// Dense GEMMs: v_wmma_f32_16x16x32_f16 with the shared weight tile staged
// through LDS via global_load_async_to_lds_b128 (ASYNCcnt double-buffer).
// Small attentions (<=16 keys decoder, 129 keys encoder) run on VALU with
// online softmax. Cross-attn over a length-1 memory collapses to a constant.
// ---------------------------------------------------------------------------

typedef __attribute__((ext_vector_type(16))) _Float16 v16h;
typedef __attribute__((ext_vector_type(8)))  _Float16 v8h;
typedef __attribute__((ext_vector_type(8)))  float    v8f;

#define DEVINL static __device__ __forceinline__

// ----- WMMA A-fragment loader (gfx1250 16x16x32 f16 layout, ISA 7.12.2) ----
// A (16x32, MxK): lanes 0-15 row M=lane, VGPR0-3 K=kb+0..7, VGPR4-7 K=kb+16..23
//                 lanes 16-31 same rows,  VGPR0-3 K=kb+8..15, VGPR4-7 K=kb+24..31
DEVINL v16h load_a_frag(const _Float16* A, long lda, int row, int kb, int half) {
  const _Float16* p = A + (size_t)row * lda + kb + half * 8;
  v8h lo = *(const v8h*)(p);
  v8h hi = *(const v8h*)(p + 16);
  v16h f;
#pragma unroll
  for (int i = 0; i < 8; ++i) { f[i] = lo[i]; f[i + 8] = hi[i]; }
  return f;
}
// B fragment from the LDS-staged tile, stored as [col(64)][k(32)] f16.
// B (32x16, KxN): lanes 0-15 K=0..7,16..23, lanes 16-31 K=8..15,24..31, column = lane&15.
DEVINL v16h load_b_lds(const _Float16* tile, int col, int half) {
  const _Float16* p = tile + col * 32 + half * 8;
  v8h lo = *(const v8h*)(p);
  v8h hi = *(const v8h*)(p + 16);
  v16h f;
#pragma unroll
  for (int i = 0; i < 8; ++i) { f[i] = lo[i]; f[i + 8] = hi[i]; }
  return f;
}

// ----- Generic GEMM:  C = act(A[MxK] @ W[NxK]^T + bias) (+resid) -----------
// Block = 256 thr = 8 waves; wave -> 16x64 strip (4 WMMA tiles, A reused).
// Weight tile (64 cols x 32 k = 4 KB) is shared by all 8 waves: staged into
// LDS with async copies, double-buffered against WMMA on the previous tile.
// NOTE: all call sites use M % 128 == 0, N % 64 == 0, K % 32 == 0.
__global__ __launch_bounds__(256) void gemm_ws(
    const _Float16* __restrict__ A, long lda,
    const _Float16* __restrict__ Wt, long ldw,
    const float* __restrict__ bias,
    const _Float16* __restrict__ resid, long ldr,
    _Float16* __restrict__ C16, long ldc16,
    float* __restrict__ C32, long ldc32,
    int M, int N, int K, int relu) {
  __shared__ __align__(16) _Float16 bsm[2][64 * 32];   // 8 KB double buffer

  const int wave = threadIdx.x >> 5;
  const int lane = threadIdx.x & 31;
  const int half = lane >> 4;
  const int nlo  = lane & 15;
  const int row0 = (blockIdx.y * 8 + wave) * 16;
  const int col0 = blockIdx.x * 64;

  // async copy helpers: 256 threads x 16B = one 64x32 f16 tile
  const int cpy_c  = threadIdx.x >> 2;                 // 0..63 (tile column)
  const int cpy_ch = threadIdx.x & 3;                  // 0..3  (16B chunk in k)
  const unsigned lds_base = (unsigned)(size_t)(&bsm[0][0]);

  const int nsteps = K >> 5;
  {  // prologue: stage tile 0
    const _Float16* src = Wt + (size_t)(col0 + cpy_c) * ldw + cpy_ch * 8;
    unsigned dst = lds_base + (unsigned)(cpy_c * 64 + cpy_ch * 16);
    asm volatile("global_load_async_to_lds_b128 %0, %1, off"
                 :: "v"(dst), "v"(src) : "memory");
  }

  v8f acc0 = {}, acc1 = {}, acc2 = {}, acc3 = {};
  const int r = row0 + nlo;
  for (int s = 0; s < nsteps; ++s) {
    const int kb = s << 5;
    if (s + 1 < nsteps) {  // stage tile s+1 into the other buffer
      const _Float16* src = Wt + (size_t)(col0 + cpy_c) * ldw + (kb + 32) + cpy_ch * 8;
      unsigned dst = lds_base + (unsigned)(((s + 1) & 1) * 4096 + cpy_c * 64 + cpy_ch * 16);
      asm volatile("global_load_async_to_lds_b128 %0, %1, off"
                   :: "v"(dst), "v"(src) : "memory");
      asm volatile("s_wait_asynccnt 0x1" ::: "memory");   // tile s landed
    } else {
      asm volatile("s_wait_asynccnt 0x0" ::: "memory");
    }
    __syncthreads();                                      // tile s visible to all waves

    __builtin_prefetch(A + (size_t)r * lda + kb + 32, 0, 3);
    v16h a = load_a_frag(A, lda, r, kb, half);
    const _Float16* bb = &bsm[s & 1][0];
    v16h b0 = load_b_lds(bb, nlo,      half);
    v16h b1 = load_b_lds(bb, 16 + nlo, half);
    v16h b2 = load_b_lds(bb, 32 + nlo, half);
    v16h b3 = load_b_lds(bb, 48 + nlo, half);
    acc0 = __builtin_amdgcn_wmma_f32_16x16x32_f16(false, a, false, b0, (short)0, acc0, false, false);
    acc1 = __builtin_amdgcn_wmma_f32_16x16x32_f16(false, a, false, b1, (short)0, acc1, false, false);
    acc2 = __builtin_amdgcn_wmma_f32_16x16x32_f16(false, a, false, b2, (short)0, acc2, false, false);
    acc3 = __builtin_amdgcn_wmma_f32_16x16x32_f16(false, a, false, b3, (short)0, acc3, false, false);
    __syncthreads();                                      // reads done: buffer reusable
  }

  // C/D layout: lane = column (lane&15), VGPR v = row v (+8 for upper lanes)
#pragma unroll
  for (int v = 0; v < 8; ++v) {
    const int m = row0 + v + half * 8;
    float vals[4] = {acc0[v], acc1[v], acc2[v], acc3[v]};
#pragma unroll
    for (int j = 0; j < 4; ++j) {
      const int n = col0 + j * 16 + nlo;
      float val = vals[j] + bias[n];
      if (relu) val = fmaxf(val, 0.0f);
      if (resid) val += (float)resid[(size_t)m * ldr + n];
      if (C16) C16[(size_t)m * ldc16 + n] = (_Float16)val;
      if (C32) C32[(size_t)m * ldc32 + n] = val;
    }
  }
}

// ----- LayerNorm over D=256:  out = LN(a (+ b)) * g + beta  (f16 in/out) ---
__global__ __launch_bounds__(256) void ln_ws(
    const _Float16* __restrict__ a, long lda,
    const _Float16* __restrict__ b, long ldb,
    const float* __restrict__ g, const float* __restrict__ beta,
    _Float16* __restrict__ out, long ldo, int ntok) {
  const int tok = blockIdx.x;
  if (tok >= ntok) return;
  const int d = threadIdx.x;
  float v = (float)a[(size_t)tok * lda + d];
  if (b) v += (float)b[(size_t)tok * ldb + d];
  __shared__ float red[256];
  red[d] = v; __syncthreads();
  for (int s = 128; s > 0; s >>= 1) { if (d < s) red[d] += red[d + s]; __syncthreads(); }
  const float mean = red[0] * (1.0f / 256.0f);
  __syncthreads();
  const float c = v - mean;
  red[d] = c * c; __syncthreads();
  for (int s = 128; s > 0; s >>= 1) { if (d < s) red[d] += red[d + s]; __syncthreads(); }
  const float var = red[0] * (1.0f / 256.0f);
  const float y = c * rsqrtf(var + 1e-5f) * g[d] + beta[d];
  out[(size_t)tok * ldo + d] = (_Float16)y;
}

// ----- FiLM modulation: shifted = tanh(gamma*x + beta) ---------------------
__global__ __launch_bounds__(256) void film_ws(
    const float* __restrict__ emb, const float* __restrict__ feat,
    const float* __restrict__ se_w, const float* __restrict__ se_b,
    float* __restrict__ shifted) {
  const int tok = blockIdx.x;       // 0..2047
  const int d   = threadIdx.x;      // 0..255
  const float* f = feat + (size_t)tok * 7;
  float ga = se_b[d], be = se_b[256 + d];
#pragma unroll
  for (int j = 0; j < 7; ++j) {
    ga = fmaf(f[j], se_w[d * 7 + j], ga);
    be = fmaf(f[j], se_w[(256 + d) * 7 + j], be);
  }
  const float x = emb[(size_t)tok * 256 + d];
  shifted[(size_t)tok * 256 + d] = tanhf(fmaf(ga, x, be));
}

// ----- real(ifft) over agent axis (N=128 cosine matmul) --------------------
__global__ __launch_bounds__(256) void ifft_real_ws(
    const float* __restrict__ shifted, _Float16* __restrict__ tokens16,
    _Float16* __restrict__ mem16) {
  const int n = blockIdx.x;         // 0..127  output agent index
  const int b = blockIdx.y;         // 0..15
  const int d = threadIdx.x;        // 0..255
  __shared__ float ct[128];
  if (d < 128) ct[d] = cosf((float)((n * d) & 127) * (6.283185307179586f / 128.0f));
  __syncthreads();
  const float* src = shifted + ((size_t)b * 128) * 256 + d;
  float acc = 0.0f;
  for (int k = 0; k < 128; ++k) acc = fmaf(ct[k], src[(size_t)k * 256], acc);
  acc *= (1.0f / 128.0f);
  const size_t seq = (size_t)b * 128 + n;
  mem16[seq * 256 + d] = (_Float16)acc;
  tokens16[(seq * 16) * 256 + d] = (_Float16)acc;   // token 0 = mem
}

// ----- Decoder incremental self-attn step (8 heads, dh=32, <=16 keys) ------
__global__ __launch_bounds__(256) void dec_attn_ws(
    const _Float16* __restrict__ qkv, _Float16* __restrict__ Kc,
    _Float16* __restrict__ Vc, _Float16* __restrict__ out, int t) {
  const int seq = blockIdx.x;       // 0..2047
  const int tid = threadIdx.x;
  const _Float16* qk = qkv + (size_t)seq * 768;
  const size_t cbase = ((size_t)seq * 16 + (t - 1)) * 256;
  Kc[cbase + tid] = qk[256 + tid];  // append k,v of token t-1
  Vc[cbase + tid] = qk[512 + tid];
  __syncthreads();
  const int head = tid >> 5, lane = tid & 31;     // one wave per head
  const int dc = head * 32 + lane;
  const float q = (float)qk[dc];
  float sc[16];
  const float scale = 0.17677669529663687f;       // 1/sqrt(32)
  for (int j = 0; j < t; ++j) {
    float p = q * (float)Kc[((size_t)seq * 16 + j) * 256 + dc];
#pragma unroll
    for (int off = 16; off > 0; off >>= 1) p += __shfl_xor(p, off, 32);
    sc[j] = p * scale;
  }
  float m = -1e30f;
  for (int j = 0; j < t; ++j) m = fmaxf(m, sc[j]);
  float den = 0.0f;
  for (int j = 0; j < t; ++j) { sc[j] = expf(sc[j] - m); den += sc[j]; }
  const float inv = 1.0f / den;
  float o = 0.0f;
  for (int j = 0; j < t; ++j)
    o = fmaf(sc[j] * inv, (float)Vc[((size_t)seq * 16 + j) * 256 + dc], o);
  out[(size_t)seq * 256 + dc] = (_Float16)o;
}

// ----- Encoder attention (4 heads, dh=64, S=129, key-pad mask) -------------
__global__ __launch_bounds__(256) void enc_attn_ws(
    const _Float16* __restrict__ qkv, const int* __restrict__ mask,
    _Float16* __restrict__ out) {
  const int bt = blockIdx.x;        // 0..255 (b*16+t)
  const int head = blockIdx.y;      // 0..3
  __shared__ int pad[129];          // 1 = masked
  const int tid = threadIdx.x;
  if (tid < 129) pad[tid] = (tid == 128) ? 0 : (mask[(size_t)bt * 128 + tid] == 0);
  __syncthreads();
  if (tid >= 129) return;           // one query per thread
  const size_t rowb = (size_t)bt * 129;
  const _Float16* qr = qkv + (rowb + tid) * 768 + head * 64;
  float q[64];
#pragma unroll
  for (int i = 0; i < 64; ++i) q[i] = (float)qr[i];
  float m = -1e30f, den = 0.0f, acc[64];
#pragma unroll
  for (int i = 0; i < 64; ++i) acc[i] = 0.0f;
  for (int j = 0; j < 129; ++j) {
    const _Float16* kr = qkv + (rowb + j) * 768 + 256 + head * 64;
    float s = 0.0f;
#pragma unroll
    for (int i = 0; i < 64; ++i) s = fmaf(q[i], (float)kr[i], s);
    s = pad[j] ? -1e9f : s * 0.125f;             // scale = 1/sqrt(64)
    const float mn = fmaxf(m, s);
    const float corr = expf(m - mn);
    const float e = expf(s - mn);
    den = den * corr + e;
    const _Float16* vr = qkv + (rowb + j) * 768 + 512 + head * 64;
#pragma unroll
    for (int i = 0; i < 64; ++i) acc[i] = fmaf(acc[i], corr, e * (float)vr[i]);
    m = mn;
  }
  const float inv = 1.0f / den;
  _Float16* orow = out + (rowb + tid) * 256 + head * 64;
#pragma unroll
  for (int i = 0; i < 64; ++i) orow[i] = (_Float16)(acc[i] * inv);
}

// ----- topo build: (bz,T,sl+1,d) from generated tokens + goal --------------
__global__ __launch_bounds__(256) void build_topo_ws(
    const _Float16* __restrict__ tokens, const float* __restrict__ goal,
    _Float16* __restrict__ topo) {
  const int row = blockIdx.x;       // 0..33023
  const int d = threadIdx.x;
  const int s = row % 129;
  const int bt = row / 129;
  const int t = bt & 15, b = bt >> 4;
  _Float16 v;
  if (s < 128) v = tokens[(((size_t)(b * 128 + s)) * 16 + t) * 256 + d];
  else         v = (_Float16)goal[(size_t)b * 256 + d];
  topo[(size_t)row * 256 + d] = v;
}

__global__ __launch_bounds__(256) void f32_to_f16_ws(
    const float* __restrict__ s, _Float16* __restrict__ d, int n) {
  const int i = blockIdx.x * 256 + threadIdx.x;
  if (i < n) d[i] = (_Float16)s[i];
}

// ---------------------------------------------------------------------------
static void launch_gemm(hipStream_t st, const _Float16* A, long lda,
                        const _Float16* Wt, long ldw, const float* bias,
                        const _Float16* resid, long ldr, _Float16* C16,
                        long ldc16, float* C32, long ldc32, int M, int N,
                        int K, int relu) {
  dim3 g(N / 64, (M + 127) / 128);
  gemm_ws<<<g, 256, 0, st>>>(A, lda, Wt, ldw, bias, resid, ldr, C16, ldc16,
                             C32, ldc32, M, N, K, relu);
}

extern "C" void kernel_launch(void* const* d_in, const int* in_sizes, int n_in,
                              void* d_out, int out_size, void* d_ws, size_t ws_size,
                              hipStream_t stream) {
  (void)n_in; (void)out_size; (void)ws_size;
  // ---- resolve input ordering (dict-insertion vs tree_leaves/sorted) ----
  const float *agent_emb, *agent_feature, *goal_emb;
  const int* agent_mask;
  const float *se_w, *se_b, *d_sa_wi, *d_sa_bi, *d_sa_wo, *d_sa_bo,
      *d_ca_wi, *d_ca_bi, *d_ca_wo, *d_ca_bo, *d_l1_w, *d_l1_b, *d_l2_w,
      *d_l2_b, *d_n1_g, *d_n1_b, *d_n2_g, *d_n2_b, *d_n3_g, *d_n3_b, *on_g,
      *on_b, *e_sa_wi, *e_sa_bi, *e_sa_wo, *e_sa_bo, *e_l1_w, *e_l1_b,
      *e_l2_w, *e_l2_b, *e_n1_g, *e_n1_b, *e_n2_g, *e_n2_b;
  agent_emb = (const float*)d_in[0];
  agent_feature = (const float*)d_in[1];
  if (in_sizes[2] == 4096) {  // insertion order
    goal_emb = (const float*)d_in[2]; agent_mask = (const int*)d_in[3];
    se_w=(const float*)d_in[4]; se_b=(const float*)d_in[5];
    d_sa_wi=(const float*)d_in[6]; d_sa_bi=(const float*)d_in[7];
    d_sa_wo=(const float*)d_in[8]; d_sa_bo=(const float*)d_in[9];
    d_ca_wi=(const float*)d_in[10]; d_ca_bi=(const float*)d_in[11];
    d_ca_wo=(const float*)d_in[12]; d_ca_bo=(const float*)d_in[13];
    d_l1_w=(const float*)d_in[14]; d_l1_b=(const float*)d_in[15];
    d_l2_w=(const float*)d_in[16]; d_l2_b=(const float*)d_in[17];
    d_n1_g=(const float*)d_in[18]; d_n1_b=(const float*)d_in[19];
    d_n2_g=(const float*)d_in[20]; d_n2_b=(const float*)d_in[21];
    d_n3_g=(const float*)d_in[22]; d_n3_b=(const float*)d_in[23];
    on_g=(const float*)d_in[24]; on_b=(const float*)d_in[25];
    e_sa_wi=(const float*)d_in[26]; e_sa_bi=(const float*)d_in[27];
    e_sa_wo=(const float*)d_in[28]; e_sa_bo=(const float*)d_in[29];
    e_l1_w=(const float*)d_in[30]; e_l1_b=(const float*)d_in[31];
    e_l2_w=(const float*)d_in[32]; e_l2_b=(const float*)d_in[33];
    e_n1_g=(const float*)d_in[34]; e_n1_b=(const float*)d_in[35];
    e_n2_g=(const float*)d_in[36]; e_n2_b=(const float*)d_in[37];
  } else {                    // sorted (tree_leaves) order
    agent_mask = (const int*)d_in[2]; goal_emb = (const float*)d_in[3];
    d_ca_bi=(const float*)d_in[4]; d_ca_bo=(const float*)d_in[5];
    d_ca_wi=(const float*)d_in[6]; d_ca_wo=(const float*)d_in[7];
    d_l1_b=(const float*)d_in[8]; d_l1_w=(const float*)d_in[9];
    d_l2_b=(const float*)d_in[10]; d_l2_w=(const float*)d_in[11];
    d_n1_b=(const float*)d_in[12]; d_n1_g=(const float*)d_in[13];
    d_n2_b=(const float*)d_in[14]; d_n2_g=(const float*)d_in[15];
    d_n3_b=(const float*)d_in[16]; d_n3_g=(const float*)d_in[17];
    d_sa_bi=(const float*)d_in[18]; d_sa_bo=(const float*)d_in[19];
    d_sa_wi=(const float*)d_in[20]; d_sa_wo=(const float*)d_in[21];
    e_l1_b=(const float*)d_in[22]; e_l1_w=(const float*)d_in[23];
    e_l2_b=(const float*)d_in[24]; e_l2_w=(const float*)d_in[25];
    e_n1_b=(const float*)d_in[26]; e_n1_g=(const float*)d_in[27];
    e_n2_b=(const float*)d_in[28]; e_n2_g=(const float*)d_in[29];
    e_sa_bi=(const float*)d_in[30]; e_sa_bo=(const float*)d_in[31];
    e_sa_wi=(const float*)d_in[32]; e_sa_wo=(const float*)d_in[33];
    on_b=(const float*)d_in[34]; on_g=(const float*)d_in[35];
    se_b=(const float*)d_in[36]; se_w=(const float*)d_in[37];
  }

  // ---- workspace layout ----
  char* WS = (char*)d_ws;
  size_t cur = 0;
  auto alloc = [&](size_t bytes) { size_t o = cur; cur += (bytes + 255) & ~(size_t)255; return o; };
  auto H = [&](size_t off) { return (_Float16*)(WS + off); };

  // persistent: f16 weights
  const size_t o_dsawi = alloc(768 * 256 * 2), o_dsawo = alloc(256 * 256 * 2);
  const size_t o_dcawi = alloc(768 * 256 * 2), o_dcawo = alloc(256 * 256 * 2);
  const size_t o_dl1 = alloc(1024 * 256 * 2), o_dl2 = alloc(256 * 1024 * 2);
  const size_t o_esawi = alloc(768 * 256 * 2), o_esawo = alloc(256 * 256 * 2);
  const size_t o_el1 = alloc(256 * 256 * 2), o_el2 = alloc(256 * 256 * 2);
  // persistent: activations
  const size_t o_tok = alloc((size_t)2048 * 16 * 256 * 2);   // generated tokens f16
  const size_t o_mem = alloc((size_t)2048 * 256 * 2);
  const size_t o_ca  = alloc((size_t)2048 * 256 * 2);
  const size_t o_vm  = alloc((size_t)2048 * 256 * 2);
  // scratch union (encoder view is the largest); decoder aliases inside it
  const size_t S = cur;
  const size_t o_topo = S;                                    // 33024*256 f16
  const size_t o_ehex = o_topo + (size_t)16908288;            // eh, later ex
  const size_t o_qkve = o_ehex + (size_t)16908288;            // 33024*768 f16
  const size_t o_eh2  = o_qkve;                               // reuse after attn
  const size_t o_effh = o_qkve + (size_t)16908288;
  const size_t o_eatt = o_qkve + (size_t)50724864;
  // decoder aliases (all dead before encoder phase starts)
  const size_t o_Kc   = S;                                    // 2048*16*256 f16
  const size_t o_shft = S;                                    // f32 FiLM scratch (pre-Kc)
  const size_t o_Vc   = o_Kc + (size_t)16777216;
  const size_t o_qkvd = o_Vc + (size_t)16777216;              // 2048*768 f16
  const size_t o_att  = o_qkvd + (size_t)3145728;
  const size_t o_sa   = o_att + (size_t)1048576;
  const size_t o_x2   = o_sa + (size_t)1048576;               // also x4
  const size_t o_x3   = o_x2 + (size_t)1048576;
  const size_t o_ffh  = o_x3 + (size_t)1048576;               // 2048*1024 f16
  const size_t o_ffo  = o_ffh + (size_t)4194304;

  // ---- convert weights to f16 ----
  struct { const float* s; size_t o; int n; } conv[] = {
      {d_sa_wi, o_dsawi, 768 * 256}, {d_sa_wo, o_dsawo, 256 * 256},
      {d_ca_wi, o_dcawi, 768 * 256}, {d_ca_wo, o_dcawo, 256 * 256},
      {d_l1_w, o_dl1, 1024 * 256},   {d_l2_w, o_dl2, 256 * 1024},
      {e_sa_wi, o_esawi, 768 * 256}, {e_sa_wo, o_esawo, 256 * 256},
      {e_l1_w, o_el1, 256 * 256},    {e_l2_w, o_el2, 256 * 256}};
  for (auto& c : conv)
    f32_to_f16_ws<<<(c.n + 255) / 256, 256, 0, stream>>>(c.s, H(c.o), c.n);

  // ---- FiLM + real(ifft) -> token 0 / memory ----
  film_ws<<<2048, 256, 0, stream>>>(agent_emb, agent_feature, se_w, se_b,
                                    (float*)(WS + o_shft));
  ifft_real_ws<<<dim3(128, 16), 256, 0, stream>>>((const float*)(WS + o_shft),
                                                  H(o_tok), H(o_mem));

  // ---- cross-attn constant: proj_o(proj_v(mem)) (softmax over 1 key == 1) ----
  launch_gemm(stream, H(o_mem), 256, H(o_dcawi) + 512 * 256, 256, d_ca_bi + 512,
              nullptr, 0, H(o_vm), 256, nullptr, 0, 2048, 256, 256, 0);
  launch_gemm(stream, H(o_vm), 256, H(o_dcawo), 256, d_ca_bo, nullptr, 0,
              H(o_ca), 256, nullptr, 0, 2048, 256, 256, 0);

  // ---- autoregressive decode (KV-cached, exact vs O(T^2) re-decode) ----
  for (int t = 1; t < 16; ++t) {
    const _Float16* tokA = H(o_tok) + (size_t)(t - 1) * 256;  // lda = 16*256
    launch_gemm(stream, tokA, 4096, H(o_dsawi), 256, d_sa_bi, nullptr, 0,
                H(o_qkvd), 768, nullptr, 0, 2048, 768, 256, 0);
    dec_attn_ws<<<2048, 256, 0, stream>>>(H(o_qkvd), H(o_Kc), H(o_Vc), H(o_att), t);
    launch_gemm(stream, H(o_att), 256, H(o_dsawo), 256, d_sa_bo, nullptr, 0,
                H(o_sa), 256, nullptr, 0, 2048, 256, 256, 0);
    ln_ws<<<2048, 256, 0, stream>>>(tokA, 4096, H(o_sa), 256, d_n1_g, d_n1_b,
                                    H(o_x2), 256, 2048);
    ln_ws<<<2048, 256, 0, stream>>>(H(o_x2), 256, H(o_ca), 256, d_n2_g, d_n2_b,
                                    H(o_x3), 256, 2048);
    launch_gemm(stream, H(o_x3), 256, H(o_dl1), 256, d_l1_b, nullptr, 0,
                H(o_ffh), 1024, nullptr, 0, 2048, 1024, 256, 1);
    launch_gemm(stream, H(o_ffh), 1024, H(o_dl2), 1024, d_l2_b, nullptr, 0,
                H(o_ffo), 256, nullptr, 0, 2048, 256, 1024, 0);
    ln_ws<<<2048, 256, 0, stream>>>(H(o_x3), 256, H(o_ffo), 256, d_n3_g, d_n3_b,
                                    H(o_x2), 256, 2048);        // x4 reuses x2
    ln_ws<<<2048, 256, 0, stream>>>(H(o_x2), 256, nullptr, 0, on_g, on_b,
                                    H(o_tok) + (size_t)t * 256, 4096, 2048);
  }

  // ---- encoder (pre-norm) over (bz*T, 129, 256) ----
  const int ME = 33024;  // 16*16*129, multiple of 16 and 128
  build_topo_ws<<<ME, 256, 0, stream>>>(H(o_tok), goal_emb, H(o_topo));
  ln_ws<<<ME, 256, 0, stream>>>(H(o_topo), 256, nullptr, 0, e_n1_g, e_n1_b,
                                H(o_ehex), 256, ME);
  launch_gemm(stream, H(o_ehex), 256, H(o_esawi), 256, e_sa_bi, nullptr, 0,
              H(o_qkve), 768, nullptr, 0, ME, 768, 256, 0);
  enc_attn_ws<<<dim3(256, 4), 256, 0, stream>>>(H(o_qkve), agent_mask, H(o_eatt));
  launch_gemm(stream, H(o_eatt), 256, H(o_esawo), 256, e_sa_bo, H(o_topo), 256,
              H(o_ehex), 256, nullptr, 0, ME, 256, 256, 0);    // ex = src + proj
  ln_ws<<<ME, 256, 0, stream>>>(H(o_ehex), 256, nullptr, 0, e_n2_g, e_n2_b,
                                H(o_eh2), 256, ME);
  launch_gemm(stream, H(o_eh2), 256, H(o_el1), 256, e_l1_b, nullptr, 0,
              H(o_effh), 256, nullptr, 0, ME, 256, 256, 1);
  launch_gemm(stream, H(o_effh), 256, H(o_el2), 256, e_l2_b, H(o_ehex), 256,
              nullptr, 0, (float*)d_out, 256, ME, 256, 256, 0);
}
